// Attention_32134945309382
// MI455X (gfx1250) — matrix operand order repaired and checked by reading, not compiled, not run on hardware
//
#include <hip/hip_runtime.h>

// ---------------------------------------------------------------------------
// MI455X / gfx1250 fused attention, bf16 WMMA (V_WMMA_F32_16X16X32_BF16).
// Flash-attention fusion avoids the 800MB score matrix; all GEMMs on WMMA.
// ---------------------------------------------------------------------------

typedef __attribute__((ext_vector_type(8)))  unsigned short v8us;   // 16 bytes
typedef __attribute__((ext_vector_type(16))) __bf16         v16bf;  // 8 VGPRs
typedef __attribute__((ext_vector_type(8)))  float          v8f;    // 8 VGPRs

__device__ __forceinline__ unsigned short f32_to_bf16(float f) {
  unsigned int u = __float_as_uint(f);
  u += 0x7FFFu + ((u >> 16) & 1u);            // round-to-nearest-even
  return (unsigned short)(u >> 16);
}

union ABFrag { v16bf bf; v8us us2[2]; };
union CFrag  { v8f v; float f[8]; };

__device__ __forceinline__ v8f wmma_bf16(v16bf a, v16bf b, v8f c) {
  // (neg_a, A, neg_b, B, c_mod, C, reuse_a, reuse_b)
  return __builtin_amdgcn_wmma_f32_16x16x32_bf16(false, a, false, b,
                                                 (short)0, c, false, false);
}

// A fragment (16x32 bf16, rows m0..m0+15, K k0..k0+31), base = A + m0*lda + k0.
// Per ISA: lanes 0-15 hold M=0..15 / K {0..7,16..23}; lanes 16-31 K {8..15,24..31}.
__device__ __forceinline__ v16bf load_frag_a(const unsigned short* base, int lda, int lane) {
  const int m  = lane & 15;
  const int k8 = (lane >> 4) << 3;            // 0 or 8
  const unsigned short* p = base + m * lda + k8;
  ABFrag f;
  f.us2[0] = *(const v8us*)(p);               // K = k8 .. k8+7
  f.us2[1] = *(const v8us*)(p + 16);          // K = k8+16 .. k8+23
  return f.bf;
}

// B fragment (32x16 bf16 K x N) from W stored row-major N x K, base = W + n0*ldb + k0.
// Per ISA: lanes 0-15 hold N=0..15 / K {0..7,16..23}; lanes 16-31 K {8..15,24..31}.
__device__ __forceinline__ v16bf load_frag_b(const unsigned short* base, int ldb, int lane) {
  const int n   = lane & 15;
  const int k16 = (lane >> 4) << 3;           // 0 or 8
  const unsigned short* p = base + n * ldb + k16;
  ABFrag f;
  f.us2[0] = *(const v8us*)(p);
  f.us2[1] = *(const v8us*)(p + 16);
  return f.bf;
}

// ---------------------------------------------------------------------------
// 0) elementwise f32 -> bf16
// ---------------------------------------------------------------------------
__global__ void f32_to_bf16_kernel(const float* __restrict__ in,
                                   unsigned short* __restrict__ out, int n) {
  int i = blockIdx.x * blockDim.x + threadIdx.x;
  if (i < n) out[i] = f32_to_bf16(in[i]);
}

// ---------------------------------------------------------------------------
// 1) QKV GEMM: (4096 x 768) x (2304 x 768)^T + bias.  Writes q/k per-head
//    row-major bf16 and V transposed (d-major) bf16 for the P*V GEMM.
//    Block = 256 thr = 8 waves = 32x64 output tile. Grid (128, 36).
// ---------------------------------------------------------------------------
__global__ void __launch_bounds__(256) qkv_gemm_kernel(
    const unsigned short* __restrict__ xb,
    const unsigned short* __restrict__ wb,
    const float* __restrict__ qkv_b,
    unsigned short* __restrict__ q_bf,
    unsigned short* __restrict__ k_bf,
    unsigned short* __restrict__ vT_bf) {
  const int lane = threadIdx.x & 31;
  const int wave = threadIdx.x >> 5;
  const int m0 = blockIdx.x * 32 + (wave >> 2) * 16;
  const int n0 = blockIdx.y * 64 + (wave & 3) * 16;

  CFrag acc;
#pragma unroll
  for (int v = 0; v < 8; ++v) acc.f[v] = 0.0f;

  for (int kk = 0; kk < 768; kk += 32) {
    v16bf a = load_frag_a(xb + m0 * 768 + kk, 768, lane);
    v16bf b = load_frag_b(wb + n0 * 768 + kk, 768, lane);
    acc.v = wmma_bf16(a, b, acc.v);
  }

  const int n     = n0 + (lane & 15);
  const int which = n / 768;                  // 0=q 1=k 2=v
  const int r     = n % 768;
  const int head  = r >> 6;
  const int d     = r & 63;
  const float bias = qkv_b[n];
  const int mo = m0 + ((lane >> 4) << 3);
#pragma unroll
  for (int v = 0; v < 8; ++v) {
    const int m = mo + v;
    const unsigned short val = f32_to_bf16(acc.f[v] + bias);
    if (which == 0)      q_bf[(head * 4096 + m) * 64 + d] = val;
    else if (which == 1) k_bf[(head * 4096 + m) * 64 + d] = val;
    else                 vT_bf[(head * 64 + d) * 4096 + m] = val;
  }
}

// ---------------------------------------------------------------------------
// 2) Rel-pos bias GEMMs.
//    mode 0: bias_h[b,qh,qw,kh] = sum_c q[b,qh,qw,c] * rph[qh-kh+63][c]
//            (fixed qh: rows = qw, lda = 64)
//    mode 1: bias_w[b,qh,qw,kw] = sum_c q[b,qh,qw,c] * rpw[qw-kw+63][c]
//            (fixed qw: rows = qh, lda = 4096)
//    W rows gathered with negative stride: row n -> rp[(fix - n + 63)*64].
//    Grid (12*64, 2), block 256 (8 waves, 16 tiles of 16x16 each, 2 per wave).
// ---------------------------------------------------------------------------
__global__ void __launch_bounds__(256) rel_gemm_kernel(
    const unsigned short* __restrict__ q_bf,
    const unsigned short* __restrict__ rph,
    const unsigned short* __restrict__ rpw,
    float* __restrict__ bias_h,
    float* __restrict__ bias_w) {
  const int lane = threadIdx.x & 31;
  const int wave = threadIdx.x >> 5;
  const int head = blockIdx.x >> 6;
  const int fix  = blockIdx.x & 63;
  const int mode = blockIdx.y;

  const unsigned short* rp = mode ? rpw : rph;
  const unsigned short* qbase =
      q_bf + head * 4096 * 64 + (mode ? fix * 64 : fix * 64 * 64);
  const int lda = mode ? 4096 : 64;
  float* out = mode ? bias_w : bias_h;

  for (int t = wave; t < 16; t += 8) {
    const int tm = (t >> 2) * 16;
    const int tn = (t & 3) * 16;
    CFrag acc;
#pragma unroll
    for (int v = 0; v < 8; ++v) acc.f[v] = 0.0f;

    for (int kk = 0; kk < 64; kk += 32) {
      v16bf a = load_frag_a(qbase + tm * lda + kk, lda, lane);
      // B row (tn+n) lives at rp[(fix-(tn+n)+63)*64]: negative row stride.
      v16bf b = load_frag_b(rp + (fix - tn + 63) * 64 + kk, -64, lane);
      acc.v = wmma_bf16(a, b, acc.v);
    }

    const int ncol  = tn + (lane & 15);
    const int mrow0 = tm + ((lane >> 4) << 3);
#pragma unroll
    for (int v = 0; v < 8; ++v) {
      const int m = mrow0 + v;
      int idx;
      if (mode == 0) idx = ((head * 64 + fix) * 64 + m) * 64 + ncol;  // qw=m, kh=ncol
      else           idx = ((head * 64 + m) * 64 + fix) * 64 + ncol;  // qh=m, kw=ncol
      out[idx] = acc.f[v];
    }
  }
}

// ---------------------------------------------------------------------------
// 3) Fused flash attention. Wave = 16 queries, loops keys in blocks of 32.
//    S = scale*(Q K^T) + bias_h + bias_w; online softmax; O += P V.
//    P relayout C->A through wave-private LDS. Grid 12*32, block 256.
// ---------------------------------------------------------------------------
__global__ void __launch_bounds__(256) flash_attn_kernel(
    const unsigned short* __restrict__ q_bf,
    const unsigned short* __restrict__ k_bf,
    const unsigned short* __restrict__ vT_bf,
    const float* __restrict__ bias_h,
    const float* __restrict__ bias_w,
    unsigned short* __restrict__ attn_out) {
  __shared__ __align__(16) unsigned short ldsP[8][16][32];

  const int lane = threadIdx.x & 31;
  const int wave = threadIdx.x >> 5;
  const int hi   = lane >> 4;
  const int head = blockIdx.x >> 5;
  const int q0   = (blockIdx.x & 31) * 128 + wave * 16;
  const float scale = 0.125f;                 // 64^-0.5

  const unsigned short* qh_base = q_bf + head * 4096 * 64;
  const unsigned short* kh_base = k_bf + head * 4096 * 64;
  const unsigned short* vh_base = vT_bf + head * 64 * 4096;

  // Q fragments held in registers for the whole key loop (d = 0..63).
  v16bf qa0 = load_frag_a(qh_base + q0 * 64 + 0,  64, lane);
  v16bf qa1 = load_frag_a(qh_base + q0 * 64 + 32, 64, lane);

  float m_run[8], l_run[8];
  CFrag o[4];
#pragma unroll
  for (int v = 0; v < 8; ++v) { m_run[v] = -3.0e38f; l_run[v] = 0.0f; }
#pragma unroll
  for (int dt = 0; dt < 4; ++dt)
#pragma unroll
    for (int v = 0; v < 8; ++v) o[dt].f[v] = 0.0f;

  // Per-row bias table base: ((head*64+qh)*64+qw)*64
  int bb[8];
#pragma unroll
  for (int v = 0; v < 8; ++v) {
    const int q = q0 + v + hi * 8;
    bb[v] = ((head * 64 + (q >> 6)) * 64 + (q & 63)) * 64;
  }

  CFrag zf;
#pragma unroll
  for (int v = 0; v < 8; ++v) zf.f[v] = 0.0f;

  for (int kb = 0; kb < 4096; kb += 32) {
    // ---- scores: two 16x16 tiles over keys kb..kb+15 / kb+16..kb+31 ----
    CFrag s0, s1;
    {
      v16bf b00 = load_frag_b(kh_base + (kb     ) * 64 + 0,  64, lane);
      v16bf b01 = load_frag_b(kh_base + (kb     ) * 64 + 32, 64, lane);
      v16bf b10 = load_frag_b(kh_base + (kb + 16) * 64 + 0,  64, lane);
      v16bf b11 = load_frag_b(kh_base + (kb + 16) * 64 + 32, 64, lane);
      s0.v = wmma_bf16(qa0, b00, zf.v);
      s0.v = wmma_bf16(qa1, b01, s0.v);
      s1.v = wmma_bf16(qa0, b10, zf.v);
      s1.v = wmma_bf16(qa1, b11, s1.v);
    }

    // ---- scale + decomposed rel-pos bias (kh uniform across the block) ----
    const int khr = kb >> 6;
    const int kw0 = (kb & 63) + (lane & 15);
#pragma unroll
    for (int v = 0; v < 8; ++v) {
      const float bh  = bias_h[bb[v] + khr];
      const float bw0 = bias_w[bb[v] + kw0];
      const float bw1 = bias_w[bb[v] + kw0 + 16];
      s0.f[v] = s0.f[v] * scale + bh + bw0;
      s1.f[v] = s1.f[v] * scale + bh + bw1;
    }

    // ---- online softmax (row groups = 16 lanes; masks 1,2,4,8 stay in-group)
#pragma unroll
    for (int v = 0; v < 8; ++v) {
      float t = fmaxf(s0.f[v], s1.f[v]);
      t = fmaxf(t, __shfl_xor(t, 1));
      t = fmaxf(t, __shfl_xor(t, 2));
      t = fmaxf(t, __shfl_xor(t, 4));
      t = fmaxf(t, __shfl_xor(t, 8));
      const float mn = fmaxf(m_run[v], t);
      const float al = __expf(m_run[v] - mn);
      s0.f[v] = __expf(s0.f[v] - mn);
      s1.f[v] = __expf(s1.f[v] - mn);
      float r = s0.f[v] + s1.f[v];
      r += __shfl_xor(r, 1);
      r += __shfl_xor(r, 2);
      r += __shfl_xor(r, 4);
      r += __shfl_xor(r, 8);
      l_run[v] = l_run[v] * al + r;
      m_run[v] = mn;
      o[0].f[v] *= al; o[1].f[v] *= al; o[2].f[v] *= al; o[3].f[v] *= al;
    }

    // ---- P (C layout) -> LDS -> A layout fragment ----
    {
      const int col = lane & 15;
#pragma unroll
      for (int v = 0; v < 8; ++v) {
        const int m = v + hi * 8;
        ldsP[wave][m][col]      = f32_to_bf16(s0.f[v]);
        ldsP[wave][m][col + 16] = f32_to_bf16(s1.f[v]);
      }
    }
    __syncthreads();
    ABFrag pf;
    {
      const int m  = lane & 15;
      const int k8 = hi << 3;
      pf.us2[0] = *(const v8us*)&ldsP[wave][m][k8];
      pf.us2[1] = *(const v8us*)&ldsP[wave][m][k8 + 16];
    }
    __syncthreads();

    // ---- O += P * V  (V stored d-major: contiguous keys per lane) ----
#pragma unroll
    for (int dt = 0; dt < 4; ++dt) {
      v16bf vb = load_frag_b(vh_base + (dt * 16) * 4096 + kb, 4096, lane);
      o[dt].v = wmma_bf16(pf.bf, vb, o[dt].v);
    }
  }

  // ---- normalize and write attn_out bf16 (token-major, c = head*64+d) ----
  const int col = lane & 15;
#pragma unroll
  for (int v = 0; v < 8; ++v) {
    const float inv = 1.0f / l_run[v];
    const int q = q0 + v + hi * 8;
#pragma unroll
    for (int dt = 0; dt < 4; ++dt)
      attn_out[q * 768 + head * 64 + dt * 16 + col] =
          f32_to_bf16(o[dt].f[v] * inv);
  }
}

// ---------------------------------------------------------------------------
// 4) Output projection: (4096 x 768) x (768 x 768)^T + bias -> f32 d_out.
//    Grid (128, 12), block 256.
// ---------------------------------------------------------------------------
__global__ void __launch_bounds__(256) proj_gemm_kernel(
    const unsigned short* __restrict__ ab,
    const unsigned short* __restrict__ wb,
    const float* __restrict__ pb,
    float* __restrict__ out) {
  const int lane = threadIdx.x & 31;
  const int wave = threadIdx.x >> 5;
  const int m0 = blockIdx.x * 32 + (wave >> 2) * 16;
  const int n0 = blockIdx.y * 64 + (wave & 3) * 16;

  CFrag acc;
#pragma unroll
  for (int v = 0; v < 8; ++v) acc.f[v] = 0.0f;

  for (int kk = 0; kk < 768; kk += 32) {
    v16bf a = load_frag_a(ab + m0 * 768 + kk, 768, lane);
    v16bf b = load_frag_b(wb + n0 * 768 + kk, 768, lane);
    acc.v = wmma_bf16(a, b, acc.v);
  }

  const int n = n0 + (lane & 15);
  const float bias = pb[n];
  const int mo = m0 + ((lane >> 4) << 3);
#pragma unroll
  for (int v = 0; v < 8; ++v)
    out[(mo + v) * 768 + n] = acc.f[v] + bias;
}

// ---------------------------------------------------------------------------
// Host-side launch. Workspace layout (bytes), total ~61.4 MB:
//   xb      @ 0         x bf16               6,291,456
//   wqkvb   @ 6291456   qkv_w bf16           3,538,944
//   wprojb  @ 9830400   proj_w bf16          1,179,648
//   rphb    @ 11010048  rel_pos_h bf16          16,384 (padded)
//   rpwb    @ 11026432  rel_pos_w bf16          16,384 (padded)
//   q_bf    @ 11042816  q per-head bf16      6,291,456
//   k_bf    @ 17334272  k per-head bf16      6,291,456
//   vT_bf   @ 23625728  v transposed bf16    6,291,456
//   bias_h  @ 29917184  f32                 12,582,912
//   bias_w  @ 42500096  f32                 12,582,912
//   ao_bf   @ 55083008  attn out bf16        6,291,456
// ---------------------------------------------------------------------------
extern "C" void kernel_launch(void* const* d_in, const int* in_sizes, int n_in,
                              void* d_out, int out_size, void* d_ws, size_t ws_size,
                              hipStream_t stream) {
  const float* x    = (const float*)d_in[0];
  const float* qkvw = (const float*)d_in[1];
  const float* qkvb = (const float*)d_in[2];
  const float* pw   = (const float*)d_in[3];
  const float* pb   = (const float*)d_in[4];
  const float* rph  = (const float*)d_in[5];
  const float* rpw  = (const float*)d_in[6];

  char* ws = (char*)d_ws;
  unsigned short* xb     = (unsigned short*)(ws + 0);
  unsigned short* wqkvb  = (unsigned short*)(ws + 6291456);
  unsigned short* wprojb = (unsigned short*)(ws + 9830400);
  unsigned short* rphb   = (unsigned short*)(ws + 11010048);
  unsigned short* rpwb   = (unsigned short*)(ws + 11026432);
  unsigned short* q_bf   = (unsigned short*)(ws + 11042816);
  unsigned short* k_bf   = (unsigned short*)(ws + 17334272);
  unsigned short* vT_bf  = (unsigned short*)(ws + 23625728);
  float*          bias_h = (float*)(ws + 29917184);
  float*          bias_w = (float*)(ws + 42500096);
  unsigned short* ao_bf  = (unsigned short*)(ws + 55083008);

  f32_to_bf16_kernel<<<(4096 * 768 + 255) / 256, 256, 0, stream>>>(x, xb, 4096 * 768);
  f32_to_bf16_kernel<<<(2304 * 768 + 255) / 256, 256, 0, stream>>>(qkvw, wqkvb, 2304 * 768);
  f32_to_bf16_kernel<<<(768 * 768 + 255) / 256, 256, 0, stream>>>(pw, wprojb, 768 * 768);
  f32_to_bf16_kernel<<<(127 * 64 + 255) / 256, 256, 0, stream>>>(rph, rphb, 127 * 64);
  f32_to_bf16_kernel<<<(127 * 64 + 255) / 256, 256, 0, stream>>>(rpw, rpwb, 127 * 64);

  qkv_gemm_kernel<<<dim3(128, 36), 256, 0, stream>>>(xb, wqkvb, qkvb, q_bf, k_bf, vT_bf);
  rel_gemm_kernel<<<dim3(12 * 64, 2), 256, 0, stream>>>(q_bf, rphb, rpwb, bias_h, bias_w);
  flash_attn_kernel<<<dim3(12 * 32), 256, 0, stream>>>(q_bf, k_bf, vT_bf, bias_h, bias_w, ao_bf);
  proj_gemm_kernel<<<dim3(128, 12), 256, 0, stream>>>(ao_bf, wprojb, pb, (float*)d_out);

  (void)in_sizes; (void)n_in; (void)out_size; (void)ws_size;
}
